// LSTM_8598524527020
// MI455X (gfx1250) — matrix operand order repaired and checked by reading, not compiled, not run on hardware
//
#include <hip/hip_runtime.h>
#include <hip/hip_bf16.h>

typedef __attribute__((ext_vector_type(16))) _Float16 v16h;
typedef __attribute__((ext_vector_type(8)))  _Float16 v8h;
typedef __attribute__((ext_vector_type(4)))  _Float16 v4h;
typedef __attribute__((ext_vector_type(8)))  float    v8f;

#define S_LEN 65536
#define HID   128
#define NG    512   // 4*HID

#define LOG2E 1.44269504088896340736f

__device__ __forceinline__ float fast_exp2(float x) {
    return __builtin_amdgcn_exp2f(x);                            // v_exp_f32
}
__device__ __forceinline__ float fast_sigmoid(float x) {
    return __builtin_amdgcn_rcpf(1.0f + fast_exp2(-LOG2E * x));
}
__device__ __forceinline__ float fast_tanh(float x) {
    // tanh(x) = 2/(1+exp(-2x)) - 1
    return fmaf(2.0f,
                __builtin_amdgcn_rcpf(1.0f + fast_exp2(-2.0f * LOG2E * x)),
                -1.0f);
}

// -------------------------------------------------------------------------
// Kernel 1: persistent single-workgroup LSTM scan. 256 threads = 8 wave32.
// Wave w owns ALL FOUR gate tiles (i,f,g,o) of hidden slice [16w, 16w+16):
// gate rows 128*s + 16w + n, s = 0..3. After the four 4-deep WMMA chains,
// lane n holds i,f,g,o for hidden unit 16w+n in registers, so the c/h
// update is fused locally: c lives in a register, no gate exchange through
// LDS, and only ONE barrier per step. h is double-buffered in LDS
// (read hbuf[t&1], write hbuf[(t+1)&1]) to avoid a read/write barrier.
// WMMA trick: A's 16 rows are all replicated with the h chunk, so every
// row of D equals the matvec -> lane n reads acc[0] directly.
// -------------------------------------------------------------------------
__launch_bounds__(256)
__global__ void lstm_scan_kernel(const float* __restrict__ x,
                                 const float* __restrict__ h0,
                                 const float* __restrict__ c0,
                                 const float* __restrict__ W_ih,
                                 const float* __restrict__ W_hh,
                                 const float* __restrict__ b_ih,
                                 const float* __restrict__ b_hh,
                                 _Float16* __restrict__ hs_out)  // (S, 128) f16
{
    __shared__ __align__(16) _Float16 hbuf[2][HID];  // double-buffered h (f16)

    const int tid    = threadIdx.x;
    const int wave   = tid >> 5;
    const int lane   = tid & 31;
    const int col    = lane & 15;   // N column within 16x16 tile
    const int hiHalf = lane >> 4;   // lane group 0: lanes0-15, 1: lanes16-31

    // ---- init state ----
    if (tid < HID) hbuf[0][tid] = (_Float16)h0[tid];
    const int hidx = 16 * wave + col;          // this lane's hidden unit
    float creg = c0[hidx];                     // cell state in register

    // ---- pack W_hh into WMMA B-operand layout (f16), per-lane gathers ----
    // B chunk kc is 32x16 (KxN): lane holds column n = lane&15,
    // K rows 32*kc + 16*hiHalf + e (e = 0..15, packed 2 halves per VGPR).
    // gates[r] = sum_k h[k] * W_hh[r,k]  ->  B[k,n] = W_hh[row_s(n), k].
    v16h  bmat[4][4];
    float wih0[4], wih1[4], bias[4];
#pragma unroll
    for (int s = 0; s < 4; ++s) {              // s: 0=i, 1=f, 2=g, 3=o
        const int row = 128 * s + hidx;        // gate row in [0, 512)
        const float* wrow = W_hh + row * HID;
#pragma unroll
        for (int kc = 0; kc < 4; ++kc) {
            const int koff = 32 * kc + 16 * hiHalf;
            v16h b;
#pragma unroll
            for (int e = 0; e < 16; ++e) b[e] = (_Float16)wrow[koff + e];
            bmat[s][kc] = b;
        }
        wih0[s] = W_ih[row * 2 + 0];           // fused input projection (I=2)
        wih1[s] = W_ih[row * 2 + 1];
        bias[s] = b_ih[row] + b_hh[row];
    }
    __syncthreads();

    const int sel = 16 * hiHalf;  // A operand: lanes16-31 hold K+16 sub-chunk

    // A chunk loader: 16 halves of h (replicated across all 16 rows of A).
    auto loadA = [&](const _Float16* hr, int kc) -> v16h {
        const v8h lo = *(const v8h*)&hr[32 * kc + sel];
        const v8h hi = *(const v8h*)&hr[32 * kc + 8 + sel];
        return __builtin_shufflevector(lo, hi, 0, 1, 2, 3, 4, 5, 6, 7,
                                               8, 9, 10, 11, 12, 13, 14, 15);
    };

    // x prefetch pipeline (uniform scalar loads, one step ahead)
    float x0 = x[0];
    float x1 = x[1];

#pragma unroll 1
    for (int t = 0; t < S_LEN; ++t) {
        // prefetch next step's x now; consumed at t+1 (hides KM latency)
        const int tn = (t + 1 < S_LEN) ? (t + 1) : t;
        const float nx0 = x[2 * tn + 0];
        const float nx1 = x[2 * tn + 1];

        const _Float16* hr = hbuf[t & 1];          // read buffer
        _Float16*       hw = hbuf[(t + 1) & 1];    // write buffer

        // -- four independent 4-deep WMMA chains, A pipelined one chunk --
        v8f acc[4] = {{}, {}, {}, {}};
        v16h a_cur = loadA(hr, 0);
#pragma unroll
        for (int kc = 0; kc < 4; ++kc) {
            v16h a_nxt;
            if (kc < 3) a_nxt = loadA(hr, kc + 1);
#pragma unroll
            for (int s = 0; s < 4; ++s)
                acc[s] = __builtin_amdgcn_wmma_f32_16x16x32_f16(
                    false, a_cur, false, bmat[s][kc], (short)0, acc[s],
                    false, false);
            if (kc < 3) a_cur = a_nxt;
        }

        // gate pre-activations: matvec + fused input projection + bias
        float pre[4];
#pragma unroll
        for (int s = 0; s < 4; ++s)
            pre[s] = acc[s][0] + fmaf(x1, wih1[s], fmaf(x0, wih0[s], bias[s]));

        // i,f,o: sigmoid; g: tanh (compile-time selection, no branches)
        const float ig = fast_sigmoid(pre[0]);
        const float fg = fast_sigmoid(pre[1]);
        const float gg = fast_tanh(pre[2]);
        const float og = fast_sigmoid(pre[3]);

        // fused c/h update in registers (lanes 16-31 compute duplicates)
        const float cn = fmaf(fg, creg, ig * gg);
        creg = cn;
        const float hn = og * fast_tanh(cn);

        if (lane < 16) {
            hw[hidx] = (_Float16)hn;               // next step's A source
            hs_out[t * HID + hidx] = (_Float16)hn; // stream for pass 2
        }
        x0 = nx0;
        x1 = nx1;
        __syncthreads();                           // single barrier per step
    }
}

// -------------------------------------------------------------------------
// Kernel 2: out[t] = tanh(h_t) . W_out + b_out. One wave per row; each lane
// covers 4 hidden units; __shfl_xor wave32 reduction. Memory-bound: 16 MB
// of f16 hs at 23.3 TB/s ~ 0.7 us.
// -------------------------------------------------------------------------
__launch_bounds__(256)
__global__ void lstm_out_kernel(const _Float16* __restrict__ hs,   // (S,128)
                                const float* __restrict__ W_out,   // (128)
                                const float* __restrict__ b_out,   // (1)
                                float* __restrict__ out)           // (S)
{
    const int lane = threadIdx.x & 31;
    const int row  = blockIdx.x * (blockDim.x >> 5) + (threadIdx.x >> 5);
    const int j0   = 4 * lane;

    const v4h hv = *(const v4h*)&hs[row * HID + j0];
    float s = 0.0f;
#pragma unroll
    for (int e = 0; e < 4; ++e)
        s += fast_tanh((float)hv[e]) * W_out[j0 + e];
#pragma unroll
    for (int off = 16; off >= 1; off >>= 1)
        s += __shfl_xor(s, off, 32);
    if (lane == 0) out[row] = s + b_out[0];
}

extern "C" void kernel_launch(void* const* d_in, const int* in_sizes, int n_in,
                              void* d_out, int out_size, void* d_ws, size_t ws_size,
                              hipStream_t stream) {
    const float* x    = (const float*)d_in[0];
    const float* h0   = (const float*)d_in[1];
    const float* c0   = (const float*)d_in[2];
    const float* Wih  = (const float*)d_in[3];
    const float* Whh  = (const float*)d_in[4];
    const float* bih  = (const float*)d_in[5];
    const float* bhh  = (const float*)d_in[6];
    const float* Wout = (const float*)d_in[7];
    const float* bout = (const float*)d_in[8];
    float* out = (float*)d_out;

    _Float16* hs = (_Float16*)d_ws;  // 65536*128*2B = 16 MB scratch

    lstm_scan_kernel<<<1, 256, 0, stream>>>(x, h0, c0, Wih, Whh, bih, bhh, hs);
    lstm_out_kernel<<<S_LEN / 8, 256, 0, stream>>>(hs, Wout, bout, out);
}
